// GraphSAGE_64647847740120
// MI455X (gfx1250) — matrix operand order repaired and checked by reading, not compiled, not run on hardware
//
#include <hip/hip_runtime.h>
#include <stdint.h>

// ---------------------------------------------------------------------------
// GraphSAGE (3x SAGEConv, mean aggregation) for gfx1250 / MI455X.
//   layer: h = relu_dropout( mean_agg(h_prev) @ Wl + h_prev @ Wr + b )
// Pipeline per layer:
//   1) zero agg (f32)                         - bandwidth kernel
//   2) edge scatter: atomicAdd f32 into agg   - resolves in 192MB L2
//   3) mn = bf16(agg * invcnt)                - bandwidth kernel
//   4) WMMA GEMM: [mn | h_prev](16x256) @ [Wl;Wr](256xd_out) + bias (+relu/drop)
// All features stored bf16. A sched_barrier(0) between the A-fragment load
// phase and the WMMA chain forces all 16 b128 loads to stay hoisted with
// distinct destination registers (single clause, one wait) instead of the
// scheduler's load->wait0->wmma serialization. Next tile is prefetched.
// ---------------------------------------------------------------------------

typedef __attribute__((ext_vector_type(16))) __bf16 v16bf;
typedef __attribute__((ext_vector_type(8)))  float  v8f;

__device__ __forceinline__ uint32_t hash32(uint32_t x) {
  x ^= x >> 16; x *= 0x7feb352dU;
  x ^= x >> 15; x *= 0x846ca68bU;
  x ^= x >> 16; return x;
}

// ----------------------------- utility kernels -----------------------------

__global__ void fill_zero_f32(float* __restrict__ p, long n) {
  long i = (long)blockIdx.x * blockDim.x + threadIdx.x;
  const long stride = (long)gridDim.x * blockDim.x;
  for (; i < n; i += stride) p[i] = 0.0f;
}

__global__ void edge_count(const long long* __restrict__ dst,
                           float* __restrict__ cnt, int n_edges) {
  int e = blockIdx.x * blockDim.x + threadIdx.x;
  if (e < n_edges) atomicAdd(cnt + (int)dst[e], 1.0f);
}

__global__ void make_invcnt(const float* __restrict__ cnt,
                            float* __restrict__ inv, int n) {
  int i = blockIdx.x * blockDim.x + threadIdx.x;
  if (i < n) inv[i] = 1.0f / fmaxf(cnt[i], 1.0f);
}

// f32 -> bf16 bulk convert (for input features x), 4 elems/thread
__global__ void cvt_f32_bf16(const float* __restrict__ in,
                             __bf16* __restrict__ out, long n4) {
  long i = (long)blockIdx.x * blockDim.x + threadIdx.x;
  if (i >= n4) return;
  const float4 v = *(const float4*)(in + i * 4);
  __bf16* o = out + i * 4;
  o[0] = (__bf16)v.x; o[1] = (__bf16)v.y;
  o[2] = (__bf16)v.z; o[3] = (__bf16)v.w;
}

// mn = bf16(agg * invcnt[row]); 4 channels/thread, 32 groups per 128-ch row
__global__ void norm_bf16(const float* __restrict__ agg,
                          const float* __restrict__ invcnt,
                          __bf16* __restrict__ mn, long n4) {
  long i = (long)blockIdx.x * blockDim.x + threadIdx.x;
  if (i >= n4) return;
  const float ic = invcnt[i >> 5];
  const float4 v = *(const float4*)(agg + i * 4);
  __bf16* o = mn + i * 4;
  o[0] = (__bf16)(v.x * ic); o[1] = (__bf16)(v.y * ic);
  o[2] = (__bf16)(v.z * ic); o[3] = (__bf16)(v.w * ic);
}

// --------------------------- scatter (aggregation) --------------------------
// One wave per edge; each lane handles 4 contiguous bf16 channels of 128.

__global__ void scatter_bf16(const __bf16* __restrict__ x,
                             const long long* __restrict__ src,
                             const long long* __restrict__ dst,
                             float* __restrict__ agg, int n_edges) {
  long tid = (long)blockIdx.x * blockDim.x + threadIdx.x;
  int e = (int)(tid >> 5);
  if (e >= n_edges) return;
  int c = ((int)tid & 31) * 4;
  int s = (int)src[e], d = (int)dst[e];
  const __bf16* p = x + (size_t)s * 128 + c;
  float* o = agg + (size_t)d * 128 + c;
  atomicAdd(o + 0, (float)p[0]);
  atomicAdd(o + 1, (float)p[1]);
  atomicAdd(o + 2, (float)p[2]);
  atomicAdd(o + 3, (float)p[3]);
}

// ------------------------------ dense (WMMA) -------------------------------
// out tile = [mn | prev] (16x256 bf16) @ [Wl ; Wr] (256 x d_out) + bias
// Wave w -> output columns [16w,16w+16). 8 K-steps of v_wmma_f32_16x16x32_bf16.
// A layout (16-bit, 16x32): lanes 0-15: K {0..7,16..23}; lanes 16-31:
// K {8..15,24..31}  -> two 16-byte chunks per lane per step (pure b128 loads).
// B layout (32x16): lane = column; lanes 0-15 hold K {0..7,16..23}, lanes 16-31 K {8..15,24..31}.

union AFrag {
  v16bf v;
  uint4 q[2];
};

template <bool RELU_DROP, bool OUT_BF16>
__global__ __launch_bounds__(256) void sage_dense(
    const __bf16* __restrict__ mn, const __bf16* __restrict__ prev,
    const float* __restrict__ Wl, const float* __restrict__ Wr,
    const float* __restrict__ bias, void* __restrict__ outp, int n_nodes,
    int d_out, int n_tiles, uint32_t seed) {
  const int lane = threadIdx.x & 31;
  const int wave = threadIdx.x >> 5;
  const int g    = lane >> 4;   // K-half selector
  const int ln   = lane & 15;
  const int col  = wave * 16 + ln;

  // --- B fragments: converted once, held in VGPRs across the node loop ---
  v16bf B[8];
#pragma unroll
  for (int ks = 0; ks < 8; ++ks) {
    const float* W = (ks < 4) ? Wl : Wr;
    const int kb = (ks & 3) * 32 + g * 8;    // K {0..7,16..23}+8g per group
#pragma unroll
    for (int j = 0; j < 16; ++j)
      B[ks][j] = (__bf16)W[(size_t)(kb + (j & 7) + ((j >> 3) << 4)) * d_out + col];
  }
  const float bv = bias[col];

  for (int tile = blockIdx.x; tile < n_tiles; tile += gridDim.x) {
    int m = tile * 16 + ln;
    if (m >= n_nodes) m = n_nodes - 1;        // clamp: EXEC stays all-1s
    const __bf16* rowm = mn   + (size_t)m * 128;
    const __bf16* rowp = prev + (size_t)m * 128;

    // --- prefetch next grid-stride tile (global_prefetch_b8) ---
    {
      const int t2 = tile + gridDim.x;
      int m2 = t2 * 16 + ln;
      if (m2 >= n_nodes) m2 = n_nodes - 1;
      if (t2 < n_tiles) {
        __builtin_prefetch(mn   + (size_t)m2 * 128, 0, 3);
        __builtin_prefetch(prev + (size_t)m2 * 128, 0, 3);
      }
    }

    // --- phase 1: load ALL 8 A fragments (16 x b128, one clause) ---
    AFrag a[8];
#pragma unroll
    for (int ks = 0; ks < 8; ++ks) {
      const __bf16* s = ((ks < 4) ? rowm : rowp) + (ks & 3) * 32;
      a[ks].q[0] = *(const uint4*)(s + 8 * g);        // K {0..7}+8g of window
      a[ks].q[1] = *(const uint4*)(s + 16 + 8 * g);   // K {16..23}+8g
    }

    // hard scheduling fence: no load may sink past this, no WMMA may hoist
    // above it -> all 8 fragments stay live in distinct registers.
    __builtin_amdgcn_sched_barrier(0);

    // --- phase 2: WMMA chain ---
    v8f c = {0.f, 0.f, 0.f, 0.f, 0.f, 0.f, 0.f, 0.f};
#pragma unroll
    for (int ks = 0; ks < 8; ++ks) {
      c = __builtin_amdgcn_wmma_f32_16x16x32_bf16(
          /*neg_a=*/false, a[ks].v, /*neg_b=*/false, B[ks],
          /*c_mod=*/(short)0, c, /*reuse_a=*/false, /*reuse_b=*/false);
    }

    // --- epilogue: lane holds rows (r + 8g), column `col` ---
#pragma unroll
    for (int r = 0; r < 8; ++r) {
      const int orow = tile * 16 + r + 8 * g;
      float v = c[r] + bv;
      if (RELU_DROP) {
        v = v > 0.0f ? v : 0.0f;
        const uint32_t idx = (uint32_t)orow * (uint32_t)d_out + (uint32_t)col;
        v = (hash32(seed ^ idx) & 1u) ? v * 2.0f : 0.0f;  // p=0.5, scale 2
      }
      if (orow < n_nodes) {
        if (OUT_BF16)
          ((__bf16*)outp)[(size_t)orow * d_out + col] = (__bf16)v;
        else
          ((float*)outp)[(size_t)orow * d_out + col] = v;
      }
    }
  }
}

// --------------------------------- driver ----------------------------------

extern "C" void kernel_launch(void* const* d_in, const int* in_sizes, int n_in,
                              void* d_out, int out_size, void* d_ws,
                              size_t ws_size, hipStream_t stream) {
  const float*     x   = (const float*)d_in[0];
  const long long* ei  = (const long long*)d_in[1];   // int64 edge_index [2,E]
  const float*     Wl0 = (const float*)d_in[2];
  const float*     Wr0 = (const float*)d_in[3];
  const float*     b0  = (const float*)d_in[4];
  const float*     Wl1 = (const float*)d_in[5];
  const float*     Wr1 = (const float*)d_in[6];
  const float*     b1  = (const float*)d_in[7];
  const float*     Wl2 = (const float*)d_in[8];
  const float*     Wr2 = (const float*)d_in[9];
  const float*     b2  = (const float*)d_in[10];

  const int n_nodes = in_sizes[0] / 128;
  const int n_edges = in_sizes[1] / 2;
  const long long* src = ei;
  const long long* dst = ei + n_edges;

  // workspace carve:
  //   agg (f32, atomics)  51.2MB
  //   mn  (bf16 mean)     25.6MB   (reused every layer)
  //   h0  (bf16)          25.6MB
  //   h1  (bf16)          25.6MB   (aliases xb: xb dead before h1 written)
  //   cnt/icnt             0.8MB
  char* w = (char*)d_ws;
  auto carve = [&](size_t bytes) {
    char* p = w;
    w += (bytes + 255) & ~(size_t)255;
    return (void*)p;
  };
  float*  agg  = (float*)carve((size_t)n_nodes * 128 * sizeof(float));
  __bf16* mn   = (__bf16*)carve((size_t)n_nodes * 128 * sizeof(__bf16));
  __bf16* h0   = (__bf16*)carve((size_t)n_nodes * 128 * sizeof(__bf16));
  __bf16* h1   = (__bf16*)carve((size_t)n_nodes * 128 * sizeof(__bf16));
  float*  cnt  = (float*)carve((size_t)n_nodes * sizeof(float));
  float*  icnt = (float*)carve((size_t)n_nodes * sizeof(float));
  __bf16* xb   = h1;  // alias: x-as-bf16 lives where h1 will be written later

  const int  n_tiles  = (n_nodes + 15) / 16;
  const int  nb_dense = n_tiles < 2048 ? n_tiles : 2048;
  const long n4       = (long)n_nodes * 32;   // 128-ch rows in groups of 4
  const int  nb4      = (int)((n4 + 255) / 256);
  const long scat_thr = (long)n_edges * 32;
  const int  scat_blk = (int)((scat_thr + 255) / 256);

  // ---- prep: degree counts + x -> bf16 ----
  fill_zero_f32<<<1024, 256, 0, stream>>>(cnt, (long)n_nodes);
  edge_count<<<(n_edges + 255) / 256, 256, 0, stream>>>(dst, cnt, n_edges);
  make_invcnt<<<(n_nodes + 255) / 256, 256, 0, stream>>>(cnt, icnt, n_nodes);
  cvt_f32_bf16<<<nb4, 256, 0, stream>>>(x, xb, n4);

  // ---- layer 0: xb -> h0 (relu+dropout) ----
  fill_zero_f32<<<4096, 256, 0, stream>>>(agg, (long)n_nodes * 128);
  scatter_bf16<<<scat_blk, 256, 0, stream>>>(xb, src, dst, agg, n_edges);
  norm_bf16<<<nb4, 256, 0, stream>>>(agg, icnt, mn, n4);
  sage_dense<true, true><<<nb_dense, 256, 0, stream>>>(
      mn, xb, Wl0, Wr0, b0, h0, n_nodes, 128, n_tiles, 0x243F6A88u);

  // ---- layer 1: h0 -> h1 (relu+dropout); h1 overwrites xb (now dead) ----
  fill_zero_f32<<<4096, 256, 0, stream>>>(agg, (long)n_nodes * 128);
  scatter_bf16<<<scat_blk, 256, 0, stream>>>(h0, src, dst, agg, n_edges);
  norm_bf16<<<nb4, 256, 0, stream>>>(agg, icnt, mn, n4);
  sage_dense<true, true><<<nb_dense, 256, 0, stream>>>(
      mn, h0, Wl1, Wr1, b1, h1, n_nodes, 128, n_tiles, 0x85A308D3u);

  // ---- layer 2: h1 -> out (f32, 64 ch, no activation) ----
  fill_zero_f32<<<4096, 256, 0, stream>>>(agg, (long)n_nodes * 128);
  scatter_bf16<<<scat_blk, 256, 0, stream>>>(h1, src, dst, agg, n_edges);
  norm_bf16<<<nb4, 256, 0, stream>>>(agg, icnt, mn, n4);
  sage_dense<false, false><<<nb_dense, 128, 0, stream>>>(
      mn, h1, Wl2, Wr2, b2, d_out, n_nodes, 64, n_tiles, 0u);
}
